// MaskGAT_89515708383725
// MI455X (gfx1250) — matrix operand order repaired and checked by reading, not compiled, not run on hardware
//
#include <hip/hip_runtime.h>
#include <hip/hip_bf16.h>
#include <math.h>

// ---------------- problem constants (match reference) ----------------
#define N_NODES   100000
#define E_EDGES   1600000
#define H_HEADS   8
#define D_DIM     16
#define HD        128
#define G_GRAPHS  128
#define OUT_DIM   64
#define EPS_BN    1e-3f
#define NEG_SLOPE 0.2f
#define THRESH    0.5f

typedef __attribute__((ext_vector_type(16))) __bf16 v16bf;
typedef __attribute__((ext_vector_type(8)))  float  v8f;

union BF16x16 {
  v16bf v;
  unsigned short s[16];
  unsigned int   u[8];
  uint4          q[2];
};

// float -> bf16 bits (round-to-nearest-even) via the native converter
__device__ inline unsigned short f2bf(float f) {
  __hip_bfloat16 h = __float2bfloat16(f);
  unsigned short u;
  __builtin_memcpy(&u, &h, 2);
  return u;
}

__device__ inline float leaky(float x) { return x > 0.0f ? x : NEG_SLOPE * x; }

// float atomic max via signed-max / unsigned-min trick (valid incl. -inf init)
__device__ inline void atomic_max_f32(float* addr, float val) {
  if (val >= 0.0f) atomicMax((int*)addr, __float_as_int(val));
  else             atomicMin((unsigned int*)addr, (unsigned int)__float_as_int(val));
}

// ---------------- generic fill ----------------
__global__ void fill_kernel(float* __restrict__ p, float v, int n) {
  int i = blockIdx.x * blockDim.x + threadIdx.x;
  if (i < n) p[i] = v;
}

// ---------------- GEMM: Y[n,128] = X[n,128] @ W[128,128] (bf16 WMMA, f32 acc) ----
// 256 threads = 8 waves; wave w handles rows [ (blk*8+w)*16 , +16 ), all 128 cols.
//
// W is staged in LDS *pre-swizzled into WMMA B-fragment order*:
//   Wswz[((kt*8 + nt)*32 + lane)*16 + e]  ==  bf16( W[(kbase + e)*128 + col] )
// where kbase = kt*32 + (lane>=16 ? 16 : 0), col = nt*16 + (lane&15).
// (B-fragment element e maps to K = kbase + e at fixed column, so each lane's
//  fragment is 16 contiguous ushorts -> two ds_load_b128 per fragment.)
__global__ __launch_bounds__(256) void gemm_bf16_wmma(
    const float* __restrict__ X, const float* __restrict__ W,
    float* __restrict__ Y, int nrows) {
  __shared__ unsigned short Wswz[4 * 8 * 32 * 16];   // 16384 bf16 = 32 KB

  const int tid = threadIdx.x;
  // fill: 8192 packed uint writes (2 bf16 each)
  for (int p = tid; p < 8192; p += 256) {
    const int e0   = (p & 7) * 2;        // element pair base within fragment
    const int frag = p >> 3;             // 0..1023 : ((kt*8+nt)*32 + lane)
    const int lane = frag & 31;
    const int nt   = (frag >> 5) & 7;
    const int kt   = frag >> 8;
    const int col   = nt * 16 + (lane & 15);
    const int kbase = kt * 32 + ((lane >> 4) & 1) * 16;
    const unsigned short s0 = f2bf(W[(kbase + e0)     * HD + col]);
    const unsigned short s1 = f2bf(W[(kbase + e0 + 1) * HD + col]);
    ((unsigned int*)Wswz)[p] = (unsigned int)s0 | ((unsigned int)s1 << 16);
  }
  __syncthreads();

  const int wave = tid >> 5;
  const int lane = tid & 31;
  const int m0   = (blockIdx.x * 8 + wave) * 16;
  if (m0 >= nrows) return;                   // wave-uniform: EXEC stays all-ones

  const int l15 = lane & 15;
  const int hi  = (lane >> 4) & 1;           // lane group 16..31

  v8f acc[8];
  const v8f zero = {0.f, 0.f, 0.f, 0.f, 0.f, 0.f, 0.f, 0.f};
#pragma unroll
  for (int nt = 0; nt < 8; ++nt) acc[nt] = zero;

  const float* xrow = X + (size_t)(m0 + l15) * HD;

#pragma unroll
  for (int kt = 0; kt < 4; ++kt) {
    const int kk = kt * 32;
    // A fragment: per lane, elements 0..7  = xrow[kk + hi*16 .. +7]
    //             elements 8..15 = xrow[kk + hi*16 + 8 .. +7]   (two 32B runs)
    const float4* r0 = (const float4*)(xrow + kk + hi * 16);
    const float4* r1 = (const float4*)(xrow + kk + hi * 16 + 8);
    const float4 a0 = r0[0], a1 = r0[1];
    const float4 a2 = r1[0], a3 = r1[1];
    BF16x16 a;
    a.s[0]  = f2bf(a0.x); a.s[1]  = f2bf(a0.y); a.s[2]  = f2bf(a0.z); a.s[3]  = f2bf(a0.w);
    a.s[4]  = f2bf(a1.x); a.s[5]  = f2bf(a1.y); a.s[6]  = f2bf(a1.z); a.s[7]  = f2bf(a1.w);
    a.s[8]  = f2bf(a2.x); a.s[9]  = f2bf(a2.y); a.s[10] = f2bf(a2.z); a.s[11] = f2bf(a2.w);
    a.s[12] = f2bf(a3.x); a.s[13] = f2bf(a3.y); a.s[14] = f2bf(a3.z); a.s[15] = f2bf(a3.w);

#pragma unroll
    for (int nt = 0; nt < 8; ++nt) {
      const uint4* bp = (const uint4*)&Wswz[(size_t)((kt * 8 + nt) * 32 + lane) * 16];
      BF16x16 b;
      b.q[0] = bp[0];                        // ds_load_b128
      b.q[1] = bp[1];                        // ds_load_b128
      acc[nt] = __builtin_amdgcn_wmma_f32_16x16x32_bf16(
          false, a.v, false, b.v, (short)0, acc[nt], false, false);
    }
  }

  // C/D layout: VGPR r, lanes 0-15 -> (M=r, N=lane); lanes 16-31 -> (M=r+8, N=lane-16)
#pragma unroll
  for (int nt = 0; nt < 8; ++nt) {
#pragma unroll
    for (int r = 0; r < 8; ++r) {
      Y[(size_t)(m0 + r + hi * 8) * HD + nt * 16 + l15] = acc[nt][r];
    }
  }
}

// ---------------- per-node attention projections el/er [N,H] ----------------
__global__ void attn_proj(const float* __restrict__ feat,
                          const float* __restrict__ al, const float* __restrict__ ar,
                          float* __restrict__ el, float* __restrict__ er) {
  int i = blockIdx.x * blockDim.x + threadIdx.x;
  if (i >= N_NODES * H_HEADS) return;
  int n = i >> 3, hh = i & 7;
  const float* f = feat + (size_t)n * HD + hh * D_DIM;
  const float* a = al + hh * D_DIM;
  const float* b = ar + hh * D_DIM;
  float sl = 0.f, sr = 0.f;
#pragma unroll
  for (int d = 0; d < D_DIM; ++d) { sl += f[d] * a[d]; sr += f[d] * b[d]; }
  el[i] = sl; er[i] = sr;
}

// ---------------- edge pass 1: segment max over dst ----------------
__global__ void edge_max_kernel(const int* __restrict__ src, const int* __restrict__ dst,
                                const float* __restrict__ el, const float* __restrict__ er,
                                float* __restrict__ emax) {
  int i = blockIdx.x * blockDim.x + threadIdx.x;
  if (i >= E_EDGES * H_HEADS) return;
  int e = i >> 3, hh = i & 7;
  int s = src[e], t = dst[e];
  float ev = leaky(el[s * H_HEADS + hh] + er[t * H_HEADS + hh]);
  atomic_max_f32(&emax[t * H_HEADS + hh], ev);
}

// ---------------- edge pass 2: softmax denominator ----------------
__global__ void edge_den_kernel(const int* __restrict__ src, const int* __restrict__ dst,
                                const float* __restrict__ el, const float* __restrict__ er,
                                const float* __restrict__ emax, float* __restrict__ den) {
  int i = blockIdx.x * blockDim.x + threadIdx.x;
  if (i >= E_EDGES * H_HEADS) return;
  int e = i >> 3, hh = i & 7;
  int s = src[e], t = dst[e];
  float ev = leaky(el[s * H_HEADS + hh] + er[t * H_HEADS + hh]);
  atomicAdd(&den[t * H_HEADS + hh], __expf(ev - emax[t * H_HEADS + hh]));
}

// ---------------- edge pass 3: weighted message scatter-add ----------------
__global__ void edge_msg_kernel(const int* __restrict__ src, const int* __restrict__ dst,
                                const float* __restrict__ el, const float* __restrict__ er,
                                const float* __restrict__ emax, const float* __restrict__ den,
                                const float* __restrict__ feat, float* __restrict__ out) {
  int i = blockIdx.x * blockDim.x + threadIdx.x;
  if (i >= E_EDGES * H_HEADS) return;
  int e = i >> 3, hh = i & 7;
  int s = src[e], t = dst[e];
  float ev = leaky(el[s * H_HEADS + hh] + er[t * H_HEADS + hh]);
  float a  = __expf(ev - emax[t * H_HEADS + hh]) / den[t * H_HEADS + hh];
  const float* fs = feat + (size_t)s * HD + hh * D_DIM;
  float* od = out + (size_t)t * HD + hh * D_DIM;
#pragma unroll
  for (int d = 0; d < D_DIM; ++d) atomicAdd(&od[d], a * fs[d]);
}

// ---------------- BN stage 1: per-column sum / sumsq ----------------
__global__ __launch_bounds__(256) void bn_stat(const float* __restrict__ x,
                                               float* __restrict__ sum, float* __restrict__ sq) {
  __shared__ float ssum[256], ssq[256];
  const int c = threadIdx.x & 127;
  const int half = threadIdx.x >> 7;
  float s = 0.f, q = 0.f;
  for (int n = blockIdx.x * 2 + half; n < N_NODES; n += gridDim.x * 2) {
    float v = x[(size_t)n * HD + c];
    s += v; q += v * v;
  }
  ssum[threadIdx.x] = s; ssq[threadIdx.x] = q;
  __syncthreads();
  if (threadIdx.x < 128) {
    atomicAdd(&sum[c], ssum[threadIdx.x] + ssum[threadIdx.x + 128]);
    atomicAdd(&sq[c],  ssq[threadIdx.x]  + ssq[threadIdx.x + 128]);
  }
}

// ---------------- BN stage 2: normalize + scale/shift + ReLU (+ residual) ----
// Note: the GAT bias (b0/b1) is a per-column constant added before BN; it cancels
// exactly in mean subtraction and leaves variance unchanged, so it is skipped.
__global__ void bn_apply(const float* __restrict__ x,
                         const float* __restrict__ sum, const float* __restrict__ sq,
                         const float* __restrict__ g, const float* __restrict__ be,
                         const float* __restrict__ res, float* __restrict__ y, int n_elems) {
  int i = blockIdx.x * blockDim.x + threadIdx.x;
  if (i >= n_elems) return;
  int c = i & 127;
  const float invN = 1.0f / (float)N_NODES;
  float mu  = sum[c] * invN;
  float var = sq[c] * invN - mu * mu;
  float v = (x[i] - mu) * rsqrtf(var + EPS_BN) * g[c] + be[c];
  v = fmaxf(v, 0.0f);
  if (res) v += res[i];
  y[i] = v;
}

// ---------------- sum pooling over graphs ----------------
__global__ void pool_kernel(const float* __restrict__ h2, const int* __restrict__ gid,
                            float* __restrict__ pooled) {
  int i = blockIdx.x * blockDim.x + threadIdx.x;
  if (i >= N_NODES * HD) return;
  int n = i >> 7, c = i & 127;
  atomicAdd(&pooled[gid[n] * HD + c], h2[i]);
}

// ---------------- masked linear head: [G,128] @ (W*mask)^T + b -> [G,64] ------
__global__ void head_kernel(const float* __restrict__ pooled, const float* __restrict__ Wp,
                            const float* __restrict__ bp, const float* __restrict__ maskr,
                            float* __restrict__ out) {
  int i = blockIdx.x * blockDim.x + threadIdx.x;
  if (i >= G_GRAPHS * OUT_DIM) return;
  int g = i / OUT_DIM, o = i % OUT_DIM;
  const float* p = pooled + g * HD;
  const float* w = Wp + o * HD;
  const float* m = maskr + o * HD;
  float s = bp[o];
#pragma unroll 4
  for (int c = 0; c < HD; ++c) s += p[c] * w[c] * (m[c] > THRESH ? 1.0f : 0.0f);
  out[i] = s;
}

// ====================================================================
extern "C" void kernel_launch(void* const* d_in, const int* in_sizes, int n_in,
                              void* d_out, int out_size, void* d_ws, size_t ws_size,
                              hipStream_t stream) {
  (void)in_sizes; (void)n_in; (void)out_size; (void)ws_size;

  const float* h     = (const float*)d_in[0];
  const int*   src   = (const int*)d_in[1];
  const int*   dst   = (const int*)d_in[2];
  const int*   gid   = (const int*)d_in[3];
  const float* W0    = (const float*)d_in[4];
  const float* al0   = (const float*)d_in[5];
  const float* ar0   = (const float*)d_in[6];
  /* b0 = d_in[7]  — cancels in BN, unused */
  const float* g0    = (const float*)d_in[8];
  const float* be0   = (const float*)d_in[9];
  const float* W1    = (const float*)d_in[10];
  const float* al1   = (const float*)d_in[11];
  const float* ar1   = (const float*)d_in[12];
  /* b1 = d_in[13] — cancels in BN, unused */
  const float* g1    = (const float*)d_in[14];
  const float* be1   = (const float*)d_in[15];
  const float* predW = (const float*)d_in[16];
  const float* predb = (const float*)d_in[17];
  const float* maskr = (const float*)d_in[18];
  float* out = (float*)d_out;

  // --------- carve workspace (~166 MB; ws poisoned -> everything is re-inited) ----
  float* ws = (float*)d_ws;
  size_t off = 0;
  auto take = [&](size_t n) { float* p = ws + off; off += n; return p; };
  float* feat   = take((size_t)N_NODES * HD);
  float* agg    = take((size_t)N_NODES * HD);
  float* h1     = take((size_t)N_NODES * HD);
  float* el     = take((size_t)N_NODES * H_HEADS);
  float* er     = take((size_t)N_NODES * H_HEADS);
  float* emax   = take((size_t)N_NODES * H_HEADS);
  float* den    = take((size_t)N_NODES * H_HEADS);
  float* bnsum  = take(HD);
  float* bnsq   = take(HD);
  float* pooled = take((size_t)G_GRAPHS * HD);

  const int NH   = N_NODES * H_HEADS;     // 800k
  const int NHD  = N_NODES * HD;          // 12.8M
  const int EH   = E_EDGES * H_HEADS;     // 12.8M
  const int B    = 256;
  const int gNHD = (NHD + B - 1) / B;
  const int gNH  = (NH + B - 1) / B;
  const int gEH  = (EH + B - 1) / B;
  const int gGEMM = (N_NODES / 16 + 7) / 8;  // 782 blocks (8 waves x 16 rows)

  auto run_layer = [&](const float* x_in, const float* W, const float* al,
                       const float* ar, const float* g, const float* be,
                       const float* res, float* y_out) {
    fill_kernel<<<gNHD, B, 0, stream>>>(agg, 0.0f, NHD);
    fill_kernel<<<gNH,  B, 0, stream>>>(emax, -INFINITY, NH);
    fill_kernel<<<gNH,  B, 0, stream>>>(den, 0.0f, NH);
    fill_kernel<<<1,    B, 0, stream>>>(bnsum, 0.0f, HD);
    fill_kernel<<<1,    B, 0, stream>>>(bnsq, 0.0f, HD);

    gemm_bf16_wmma<<<gGEMM, B, 0, stream>>>(x_in, W, feat, N_NODES);
    attn_proj<<<gNH, B, 0, stream>>>(feat, al, ar, el, er);
    edge_max_kernel<<<gEH, B, 0, stream>>>(src, dst, el, er, emax);
    edge_den_kernel<<<gEH, B, 0, stream>>>(src, dst, el, er, emax, den);
    edge_msg_kernel<<<gEH, B, 0, stream>>>(src, dst, el, er, emax, den, feat, agg);
    bn_stat<<<512, B, 0, stream>>>(agg, bnsum, bnsq);
    bn_apply<<<gNHD, B, 0, stream>>>(agg, bnsum, bnsq, g, be, res, y_out, NHD);
  };

  // layer 0: GAT -> BN -> ReLU                  (h  -> h1)
  run_layer(h, W0, al0, ar0, g0, be0, nullptr, h1);
  // layer 1: GAT -> BN -> ReLU + residual(h1)   (h1 -> agg, in-place over agg)
  run_layer(h1, W1, al1, ar1, g1, be1, h1, agg);

  // sum pooling + masked linear head
  fill_kernel<<<(G_GRAPHS * HD + B - 1) / B, B, 0, stream>>>(pooled, 0.0f, G_GRAPHS * HD);
  pool_kernel<<<gNHD, B, 0, stream>>>(agg, gid, pooled);
  head_kernel<<<(G_GRAPHS * OUT_DIM + B - 1) / B, B, 0, stream>>>(pooled, predW, predb, maskr, out);
}
